// GraphEncoder_58506044506643
// MI455X (gfx1250) — compile-verified, matrix operand order checked by reading, not run on hardware
//
#include <hip/hip_runtime.h>
#include <hip/hip_bf16.h>
#include <math.h>

// ---------------------------------------------------------------------------
// MI455X (gfx1250) implementation of the hierarchical GraphEncoder.
// Dense per-node projections use v_wmma_f32_16x16x32_f16 (f32 accumulate),
// fed by ds_load_b128 (LDS-staged A tile, shared by D/16 waves) and
// global_load_b128 (f16 weights); outputs stored as coalesced b128 rows.
// Irregular edge softmax/aggregation uses L2 float atomics (192MB L2).
// ---------------------------------------------------------------------------

typedef _Float16 half_t;
typedef __attribute__((ext_vector_type(16))) _Float16 v16h;
typedef __attribute__((ext_vector_type(8)))  _Float16 v8h;
typedef __attribute__((ext_vector_type(8)))  float    v8f;

#define NHI 5000
#define NLO 250000
#define KLOW 50
#define EHI 80000
#define ELO 1500000
#define CDIV(a,b) (((a)+(b)-1)/(b))

// ---------------------------------------------------------------------------
// WMMA GEMM: Y[N,D] = act(X[gidx][N,K] @ W[D,K]^T + bias[D])
// Block = dim3(32, D/16): D/16 waves, one 16-row tile per block, one 16-col
// tile per wave.  A tile is converted to f16 once into LDS and shared.
//
// CDNA5 16-bit A layout (16x32, MxK): lane L holds M=L%16; its 16 halves are
// two contiguous 8-half runs at K = k0+8*(L/16) and K = k0+16+8*(L/16) -> one
// 16-byte load each.  B (32x16, KxN) mirrors A with N in place of M; since
// B[k][n] = W[n][k] the B runs are contiguous rows of the f16 weight.
// f32 C/D layout: VGPR r of lane L holds C[r + 8*(L/16)][L%16].
// ---------------------------------------------------------------------------
template<int K, int D>
__global__ __launch_bounds__(32 * (D / 16))
void k_gemm_wmma(const float* __restrict__ X, const int* __restrict__ gidx,
                 const half_t* __restrict__ W, const float* __restrict__ bias,
                 float* __restrict__ Y, int N, int relu) {
    constexpr int NT = 32 * (D / 16);
    __shared__ half_t sA[16 * K];      // 16 x K f16 A tile
    __shared__ float  sO[16 * D];      // 16 x D f32 output tile

    const int lane   = threadIdx.x;          // 0..31
    const int wave   = threadIdx.y;          // 0..D/16-1
    const int t      = wave * 32 + lane;
    const int laneLo = lane & 15;
    const int laneHi = lane >> 4;
    const int row0   = blockIdx.x * 16;
    const int col0   = wave * 16;

    // ---- cooperative A staging: global f32 (optionally gathered) -> LDS f16
    for (int i = t; i < 16 * K; i += NT) {
        int r = i / K, k = i - r * K;
        int arow = row0 + r;
        if (arow >= N) arow = N - 1;
        int xrow = gidx ? gidx[arow] : arow;
        sA[i] = (half_t)X[(size_t)xrow * K + k];
    }
    __syncthreads();

    const half_t* wr = W + (size_t)(col0 + laneLo) * K;
    v8f acc = {};
#pragma unroll
    for (int k0 = 0; k0 < K; k0 += 32) {
        const v8h* ap = (const v8h*)(sA + laneLo * K + k0 + 8 * laneHi);  // ds_load_b128 x2
        const v8h* bp = (const v8h*)(wr + k0 + 8 * laneHi);               // global_load_b128 x2
        v16h a = __builtin_shufflevector(ap[0], ap[2], 0, 1, 2, 3, 4, 5, 6, 7,
                                         8, 9, 10, 11, 12, 13, 14, 15);
        v16h b = __builtin_shufflevector(bp[0], bp[2], 0, 1, 2, 3, 4, 5, 6, 7,
                                         8, 9, 10, 11, 12, 13, 14, 15);
        acc = __builtin_amdgcn_wmma_f32_16x16x32_f16(
                  false, a, false, b, (short)0, acc, false, false);
    }

    // ---- epilogue: bias + act into LDS, then coalesced b128 row stores
    const float bcol = bias[col0 + laneLo];
#pragma unroll
    for (int r = 0; r < 8; ++r) {
        float v = acc[r] + bcol;
        if (relu) v = fmaxf(v, 0.0f);
        sO[(r + 8 * laneHi) * D + col0 + laneLo] = v;
    }
    __syncthreads();
    for (int i = t; i < 16 * D / 4; i += NT) {
        int row = (i * 4) / D, col = (i * 4) % D;
        if (row0 + row < N)
            *(float4*)(Y + (size_t)(row0 + row) * D + col) = *(const float4*)(sO + row * D + col);
    }
}

// ---------------------------------------------------------------------------
__global__ void k_f32_to_f16(const float* __restrict__ s, half_t* __restrict__ d, int n) {
    int i = blockIdx.x * blockDim.x + threadIdx.x;
    if (i < n) d[i] = (half_t)s[i];
}
__global__ void k_fill_f32(float* p, float v, size_t n) {
    size_t i = (size_t)blockIdx.x * blockDim.x + threadIdx.x;
    if (i < n) p[i] = v;
}
__global__ void k_fill_u32(unsigned* p, unsigned v, size_t n) {
    size_t i = (size_t)blockIdx.x * blockDim.x + threadIdx.x;
    if (i < n) p[i] = v;
}
__global__ void k_add(float* __restrict__ dst, const float* __restrict__ src, size_t n) {
    size_t i = (size_t)blockIdx.x * blockDim.x + threadIdx.x;
    if (i < n) dst[i] += src[i];
}

// pe_high = cdist(X_high, X_high[anchors]); high2 = tile(X,16)+pe with
// columns feat_perm[16:32] zeroed.
__global__ void k_high_embed(const float* __restrict__ Xh, const int* __restrict__ anch,
                             const int* __restrict__ perm,
                             float* __restrict__ peh, float* __restrict__ high2, int n) {
    int idx = blockIdx.x * blockDim.x + threadIdx.x;
    if (idx >= n * 32) return;
    int node = idx >> 5, d = idx & 31;
    int a = anch[d];
    float dx = Xh[2 * node] - Xh[2 * a];
    float dy = Xh[2 * node + 1] - Xh[2 * a + 1];
    float sq = dx * dx + dy * dy;
    float dist = sq > 0.0f ? sqrtf(sq) : 0.0f;
    peh[idx] = dist;
    float he = Xh[2 * node + (d & 1)] + dist;   // tile(X_high,(1,16)) col d -> X[:,d%2]
    bool z = false;
    for (int j = 16; j < 32; ++j) if (perm[j] == d) z = true;
    high2[idx] = z ? 0.0f : he;
}

// Per-segment descending stable rank -> linspace(0,1).  Segments are fixed
// contiguous blocks of 50.  rank_i = #{j : v_j > v_i or (v_j==v_i and j<i)}
__global__ void k_ranknorm(const float* __restrict__ Xl, float* __restrict__ rank) {
    __shared__ float v[64];
    int s = blockIdx.x, t = threadIdx.x;
    if (t < KLOW) v[t] = Xl[s * KLOW + t];
    __syncthreads();
    if (t < KLOW) {
        float vi = v[t];
        int r = 0;
        for (int j = 0; j < KLOW; ++j) {
            float vj = v[j];
            if (vj > vi || (vj == vi && j < t)) r++;
        }
        rank[s * KLOW + t] = (float)r / (float)(KLOW - 1);
    }
}

// low2 = where(drop, X_low + sb*pe_low + (1-sb)*pe_high[batch], 0)
__global__ void k_low2(const float* __restrict__ Xl, const float* __restrict__ rank,
                       const float* __restrict__ peh, const int* __restrict__ batch,
                       const unsigned char* __restrict__ drop, const float* __restrict__ beta,
                       float* __restrict__ low2, int n) {
    int idx = blockIdx.x * blockDim.x + threadIdx.x;
    if (idx >= n * 32) return;
    int node = idx >> 5, d = idx & 31;
    if (!drop[node]) { low2[idx] = 0.0f; return; }
    float sb = 1.0f / (1.0f + expf(-beta[0]));
    int i = d >> 1;
    float div = expf((float)i * 2.0f * (-logf(10000.0f) / 32.0f));
    float ang = rank[node] * div;
    float pe = (d & 1) ? cosf(ang) : sinf(ang);
    float ph = peh[(size_t)batch[node] * 32 + d];
    low2[idx] = Xl[node] + sb * pe + (1.0f - sb) * ph;
}

// segment_mean over fixed contiguous blocks of 50 rows of [*, D]
__global__ void k_seg_mean(const float* __restrict__ L, float* __restrict__ G,
                           int nhigh, int D) {
    int idx = blockIdx.x * blockDim.x + threadIdx.x;
    if (idx >= nhigh * D) return;
    int h = idx / D, d = idx % D;
    const float* p = L + (size_t)h * KLOW * D + d;
    float s = 0.0f;
    for (int j = 0; j < KLOW; ++j) s += p[(size_t)j * D];
    G[idx] = s / (float)KLOW;
}

// ---------------------------------------------------------------------------
// Edge softmax.  Float segment-max via order-preserving uint encoding.
// ---------------------------------------------------------------------------
__device__ inline unsigned f2ord(float f) {
    unsigned u = __float_as_uint(f);
    return (u & 0x80000000u) ? ~u : (u | 0x80000000u);
}
__device__ inline float ord2f(unsigned u) {
    return (u & 0x80000000u) ? __uint_as_float(u & 0x7fffffffu) : __uint_as_float(~u);
}

__global__ void k_edge_score(const float* __restrict__ Q, const float* __restrict__ K,
                             const int* __restrict__ src, const int* __restrict__ dst,
                             float* __restrict__ scores, unsigned* __restrict__ m, int E) {
    int e = blockIdx.x * blockDim.x + threadIdx.x;
    if (e >= E) return;
    int s = src[e], d = dst[e];
    const float4* q4 = (const float4*)(Q + (size_t)d * 64);   // b128 loads
    const float4* k4 = (const float4*)(K + (size_t)s * 64);
#pragma unroll
    for (int h = 0; h < 4; ++h) {
        float acc = 0.0f;
#pragma unroll
        for (int j = 0; j < 4; ++j) {
            float4 qv = q4[h * 4 + j], kv = k4[h * 4 + j];
            acc += qv.x * kv.x + qv.y * kv.y + qv.z * kv.z + qv.w * kv.w;
        }
        acc *= 0.25f;                              // 1/sqrt(C), C=16
        scores[(size_t)e * 4 + h] = acc;
        atomicMax(&m[(size_t)d * 4 + h], f2ord(acc));
    }
}

// alpha = ex/denom factored out of the sum: agg = sum(ex*v), then /denom later
__global__ void k_edge_agg(const float* __restrict__ V,
                           const int* __restrict__ src, const int* __restrict__ dst,
                           const float* __restrict__ scores, const unsigned* __restrict__ m,
                           float* __restrict__ denom, float* __restrict__ agg, int E) {
    int idx = blockIdx.x * blockDim.x + threadIdx.x;
    int e = idx >> 2, h = idx & 3;
    if (e >= E) return;
    int d = dst[e], s = src[e];
    const float4* v4 = (const float4*)(V + (size_t)s * 64 + h * 16);
    __builtin_prefetch(v4, 0, 1);                  // global_prefetch_b8
    float ex = expf(scores[(size_t)e * 4 + h] - ord2f(m[(size_t)d * 4 + h]));
    atomicAdd(&denom[(size_t)d * 4 + h], ex);
    float* ar = agg + (size_t)d * 64 + h * 16;
#pragma unroll
    for (int j = 0; j < 4; ++j) {
        float4 vv = v4[j];
        atomicAdd(&ar[j * 4 + 0], ex * vv.x);
        atomicAdd(&ar[j * 4 + 1], ex * vv.y);
        atomicAdd(&ar[j * 4 + 2], ex * vv.z);
        atomicAdd(&ar[j * 4 + 3], ex * vv.w);
    }
}

// conv-out = agg/(denom+1e-16) + skip, then LayerNorm + ReLU (wave32, 2 dims/lane)
// Safe in-place (out may alias agg): each output element read exactly once.
__global__ __launch_bounds__(32)
void k_finalize(const float* __restrict__ agg, const float* __restrict__ denom,
                const float* __restrict__ S, const float* __restrict__ lnw,
                const float* __restrict__ lnb, float* __restrict__ out, int N) {
    int n = blockIdx.x, lane = threadIdx.x;
    size_t b = (size_t)n * 64;
    float x0 = agg[b + lane]      / (denom[(size_t)n * 4 + (lane >> 4)]     + 1e-16f) + S[b + lane];
    float x1 = agg[b + 32 + lane] / (denom[(size_t)n * 4 + 2 + (lane >> 4)] + 1e-16f) + S[b + 32 + lane];
    float s = x0 + x1;
    for (int off = 16; off; off >>= 1) s += __shfl_xor(s, off, 32);
    float mu = s * (1.0f / 64.0f);
    float d0 = x0 - mu, d1 = x1 - mu;
    float vv = d0 * d0 + d1 * d1;
    for (int off = 16; off; off >>= 1) vv += __shfl_xor(vv, off, 32);
    float inv = rsqrtf(vv * (1.0f / 64.0f) + 1e-5f);
    out[b + lane]      = fmaxf(d0 * inv * lnw[lane]      + lnb[lane],      0.0f);
    out[b + 32 + lane] = fmaxf(d1 * inv * lnw[32 + lane] + lnb[32 + lane], 0.0f);
}

// gated blend: s = sigmoid(A·WgA + B·WgB); out = s*A + (1-s)*B   (wave32/node)
__global__ __launch_bounds__(32)
void k_gate(const float* __restrict__ A, const float* __restrict__ B,
            const float* __restrict__ WgA, const float* __restrict__ WgB,
            float* __restrict__ out, const int* __restrict__ bidx, int N) {
    int n = blockIdx.x, lane = threadIdx.x;
    int bn = bidx ? bidx[n] : n;
    size_t ba = (size_t)n * 64, bb = (size_t)bn * 64;
    float a0 = A[ba + lane], a1 = A[ba + 32 + lane];
    float b0 = B[bb + lane], b1 = B[bb + 32 + lane];
    float dp = a0 * WgA[lane] + a1 * WgA[32 + lane] + b0 * WgB[lane] + b1 * WgB[32 + lane];
    for (int off = 16; off; off >>= 1) dp += __shfl_xor(dp, off, 32);
    float s = 1.0f / (1.0f + expf(-dp));
    out[ba + lane]      = s * a0 + (1.0f - s) * b0;
    out[ba + 32 + lane] = s * a1 + (1.0f - s) * b1;
}

// ---------------------------------------------------------------------------
// Input index map: jax pytree flatten of params (alphabetical dict keys at
// each level) followed by remaining setup_inputs entries in insertion order.
// ---------------------------------------------------------------------------
enum {
    IN_AH_WI = 0, IN_AH_WO, IN_AH_BI, IN_AH_BO,
    IN_AL_WI, IN_AL_WO, IN_AL_BI, IN_AL_BO,
    IN_BETA,
    IN_L1 = 9, IN_L2 = 21, IN_L3 = 33,          // 12 leaves each
    IN_MLPH_W = 45, IN_MLPH_B, IN_MLPL_W, IN_MLPL_B,
    IN_PROJ_W = 49, IN_PROJ_B,
    IN_XH = 51, IN_XL, IN_BATCH, IN_EH, IN_EL, IN_ANCH, IN_PERM, IN_DROP, IN_PEDIM
};
// layer leaf order (sorted keys): W_hl, W_lh, Wk, Wq, Ws, Wv, bk, bq, bs, bv, ln_b, ln_w
enum { LW_HL = 0, LW_LH, LW_K, LW_Q, LW_S, LW_V, LB_K, LB_Q, LB_S, LB_V, L_LNB, L_LNW };

extern "C" void kernel_launch(void* const* d_in, const int* in_sizes, int n_in,
                              void* d_out, int out_size, void* d_ws, size_t ws_size,
                              hipStream_t stream) {
    (void)in_sizes; (void)n_in; (void)out_size; (void)ws_size;
    auto F = [&](int i) { return (const float*)d_in[i]; };
    auto I = [&](int i) { return (const int*)d_in[i]; };

    const float* XH    = F(IN_XH);
    const float* XL    = F(IN_XL);
    const int*   batch = I(IN_BATCH);
    const int*   ehs = I(IN_EH),          * ehd = I(IN_EH) + EHI;
    const int*   els = I(IN_EL),          * eld = I(IN_EL) + ELO;
    const int*   anch = I(IN_ANCH);
    const int*   perm = I(IN_PERM);
    const unsigned char* drop = (const unsigned char*)d_in[IN_DROP];
    const float* beta = F(IN_BETA);
    float* out = (float*)d_out;

    // ---- workspace carve-up (floats) ----
    float* wsf = (float*)d_ws;
    size_t off = 0;
    auto alloc = [&](size_t n) { float* p = wsf + off; off += n; return p; };
    float* rank  = alloc(NLO);
    float* peh   = alloc((size_t)NHI * 32);
    float* high2 = alloc((size_t)NHI * 32);
    float* g     = alloc((size_t)NHI * 32);
    float* hx    = alloc((size_t)NHI * 64);
    float* ht1   = alloc((size_t)NHI * 64);
    float* ht2   = alloc((size_t)NHI * 64);
    float* hconv = alloc((size_t)NHI * 64);
    float* hb0   = alloc((size_t)NHI * 64);
    float* hb1   = alloc((size_t)NHI * 64);
    float* hb2   = alloc((size_t)NHI * 64);
    float* hb3   = alloc((size_t)NHI * 64);
    float* Lb0   = alloc((size_t)NLO * 64);
    float* Lb1   = alloc((size_t)NLO * 64);
    float* Lb2   = alloc((size_t)NLO * 64);
    float* Qb    = alloc((size_t)NLO * 64);
    float* Kb    = alloc((size_t)NLO * 64);
    float* Vb    = alloc((size_t)NLO * 64);
    float* Sb    = alloc((size_t)NLO * 64);
    float* AGG   = alloc((size_t)NLO * 64);   // also reused for low2 early on
    float* scor  = alloc((size_t)ELO * 4);
    float* denom = alloc((size_t)NLO * 4);
    unsigned* mb = (unsigned*)alloc((size_t)NLO * 4);
    half_t* w16  = (half_t*)alloc(40000);     // 80000 halves of f16 weights

    // ---- convert weights to f16 ----
    half_t* wvh = w16;          half_t* woh = w16 + 1024;
    half_t* wvl = w16 + 2048;   half_t* wol = w16 + 3072;
    half_t* mh  = w16 + 4096;   half_t* ml  = w16 + 6144;
    half_t* lw16[3][4];         // per layer: Wq, Wk, Wv, Ws
    half_t* pw  = w16 + 8192;
    for (int l = 0; l < 3; ++l)
        for (int j = 0; j < 4; ++j) { lw16[l][j] = pw; pw += 4096; }
    half_t* pj16 = pw;

    struct Cvt { const float* s; half_t* d; int n; };
    const int LB[3] = { IN_L1, IN_L2, IN_L3 };
    Cvt jobs[20];
    int nj = 0;
    jobs[nj++] = { F(IN_AH_WI) + 64 * 32, wvh, 1024 };   // Wv slice of Wi
    jobs[nj++] = { F(IN_AH_WO),           woh, 1024 };
    jobs[nj++] = { F(IN_AL_WI) + 64 * 32, wvl, 1024 };
    jobs[nj++] = { F(IN_AL_WO),           wol, 1024 };
    jobs[nj++] = { F(IN_MLPH_W), mh, 2048 };
    jobs[nj++] = { F(IN_MLPL_W), ml, 2048 };
    for (int l = 0; l < 3; ++l) {
        jobs[nj++] = { F(LB[l] + LW_Q), lw16[l][0], 4096 };
        jobs[nj++] = { F(LB[l] + LW_K), lw16[l][1], 4096 };
        jobs[nj++] = { F(LB[l] + LW_V), lw16[l][2], 4096 };
        jobs[nj++] = { F(LB[l] + LW_S), lw16[l][3], 4096 };
    }
    jobs[nj++] = { F(IN_PROJ_W), pj16, 4096 };
    for (int j = 0; j < nj; ++j)
        k_f32_to_f16<<<CDIV(jobs[j].n, 256), 256, 0, stream>>>(jobs[j].s, jobs[j].d, jobs[j].n);

    auto gemm = [&](const float* X, const int* gi, const half_t* W, const float* bias,
                    float* Y, int N, int K, int D, int relu) {
        dim3 grd(CDIV(N, 16));
        if (K == 32 && D == 32)
            k_gemm_wmma<32, 32><<<grd, dim3(32, 2), 0, stream>>>(X, gi, W, bias, Y, N, relu);
        else if (K == 32 && D == 64)
            k_gemm_wmma<32, 64><<<grd, dim3(32, 4), 0, stream>>>(X, gi, W, bias, Y, N, relu);
        else
            k_gemm_wmma<64, 64><<<grd, dim3(32, 4), 0, stream>>>(X, gi, W, bias, Y, N, relu);
    };

    // ---- positional encodings + hierarchical blending ----
    k_high_embed<<<CDIV(NHI * 32, 256), 256, 0, stream>>>(XH, anch, perm, peh, high2, NHI);
    k_ranknorm<<<NHI, 64, 0, stream>>>(XL, rank);
    k_low2<<<CDIV(NLO * 32, 256), 256, 0, stream>>>(XL, rank, peh, batch, drop, beta, AGG, NLO);
    k_seg_mean<<<CDIV(NHI * 32, 256), 256, 0, stream>>>(AGG, g, NHI, 32);

    // mha_high (seq-len-1 => out_proj(v_proj(g))) then mlp_high
    gemm(g,   nullptr, wvh, F(IN_AH_BI) + 64, ht1, NHI, 32, 32, 0);
    gemm(ht1, nullptr, woh, F(IN_AH_BO),      ht2, NHI, 32, 32, 0);
    gemm(ht2, nullptr, mh,  F(IN_MLPH_B),     hb0, NHI, 32, 64, 1);
    // mha_low on high2[batch], then mlp_low
    gemm(high2, batch,   wvl, F(IN_AL_BI) + 64, Qb,  NLO, 32, 32, 0);
    gemm(Qb,    nullptr, wol, F(IN_AL_BO),      Kb,  NLO, 32, 32, 0);
    gemm(Kb,    nullptr, ml,  F(IN_MLPL_B),     Lb0, NLO, 32, 64, 1);

    auto run_conv = [&](const float* x, int N, const int* esrc, const int* edst, int E,
                        int lb, half_t* const* lw, float* o) {
        gemm(x, nullptr, lw[0], F(lb + LB_Q), Qb, N, 64, 64, 0);
        gemm(x, nullptr, lw[1], F(lb + LB_K), Kb, N, 64, 64, 0);
        gemm(x, nullptr, lw[2], F(lb + LB_V), Vb, N, 64, 64, 0);
        gemm(x, nullptr, lw[3], F(lb + LB_S), Sb, N, 64, 64, 0);
        k_fill_u32<<<CDIV(N * 4, 256), 256, 0, stream>>>(mb, 0x007FFFFFu, (size_t)N * 4); // ord(-inf)
        k_fill_f32<<<CDIV(N * 4, 256), 256, 0, stream>>>(denom, 0.0f, (size_t)N * 4);
        k_fill_f32<<<CDIV((size_t)N * 64, 256), 256, 0, stream>>>(AGG, 0.0f, (size_t)N * 64);
        k_edge_score<<<CDIV(E, 128), 128, 0, stream>>>(Qb, Kb, esrc, edst, scor, mb, E);
        k_edge_agg<<<CDIV(E * 4, 128), 128, 0, stream>>>(Vb, esrc, edst, scor, mb, denom, AGG, E);
        k_finalize<<<N, 32, 0, stream>>>(AGG, denom, Sb, F(lb + L_LNW), F(lb + L_LNB), o, N);
    };

    auto run_layer = [&](int li, const float* hin, const float* lin, float* hout, float* lout) {
        int lb = LB[li];
        run_conv(hin, NHI, ehs, ehd, EHI, lb, lw16[li], hconv);
        run_conv(lin, NLO, els, eld, ELO, lb, lw16[li], AGG);     // l_conv in-place in AGG
        k_seg_mean<<<CDIV(NHI * 64, 256), 256, 0, stream>>>(AGG, hx, NHI, 64);
        const float* Wlh = F(lb + LW_LH);   // concat([h, x])  @ W_lh
        const float* Whl = F(lb + LW_HL);   // concat([hb, l]) @ W_hl
        k_gate<<<NHI, 32, 0, stream>>>(hconv, hx,  Wlh,      Wlh + 64, hout, nullptr, NHI);
        k_gate<<<NLO, 32, 0, stream>>>(AGG, hconv, Whl + 64, Whl,      lout, batch,   NLO);
    };

    run_layer(0, hb0, Lb0, hb1, Lb1);
    run_layer(1, hb1, Lb1, hb2, Lb2);
    k_add<<<CDIV((size_t)NHI * 64, 256), 256, 0, stream>>>(hb2, hb1, (size_t)NHI * 64);
    k_add<<<CDIV((size_t)NLO * 64, 256), 256, 0, stream>>>(Lb2, Lb1, (size_t)NLO * 64);
    run_layer(2, hb2, Lb2, hb3, Lb0);                              // l3 into Lb0
    k_add<<<CDIV((size_t)NHI * 64, 256), 256, 0, stream>>>(hb3, hb2, (size_t)NHI * 64);
    k_add<<<CDIV((size_t)NLO * 64, 256), 256, 0, stream>>>(Lb0, Lb2, (size_t)NLO * 64);

    // final projection + relu -> (proj(h3), proj(l3)) concatenated
    gemm(hb3, nullptr, pj16, F(IN_PROJ_B), out,                    NHI, 64, 64, 1);
    gemm(Lb0, nullptr, pj16, F(IN_PROJ_B), out + (size_t)NHI * 64, NLO, 64, 64, 1);
}
